// Net_12386685682286
// MI455X (gfx1250) — matrix operand order repaired and checked by reading, not compiled, not run on hardware
//
#include <hip/hip_runtime.h>
#include <hip/hip_bf16.h>
#include <math.h>

#define NN      100000
#define HID     256
#define NLAYER  4
#define HEADS   8
#define BASES   4
#define AGGS    3
#define DH      32
#define BF      128
#define NG      5000
#define VOCAB   128
#define NFEAT   9
#define COMBD   (HEADS*BASES*AGGS)   // 96
#define EPSV    1e-5f

typedef __attribute__((ext_vector_type(16))) __bf16 v16bf;
typedef __attribute__((ext_vector_type(8)))  float  v8f;

// ---------------------------------------------------------------- utilities
__global__ void k_fill(float* p, size_t n, float v) {
    size_t i = (size_t)blockIdx.x * blockDim.x + threadIdx.x;
    size_t stride = (size_t)gridDim.x * blockDim.x;
    for (; i < n; i += stride) p[i] = v;
}

__device__ __forceinline__ void atomicMaxF(float* addr, float val) {
    if (val >= 0.0f) atomicMax((int*)addr, __float_as_int(val));
    else             atomicMin((unsigned int*)addr, __float_as_uint(val));
}

// ---------------------------------------------------------------- encoder
__global__ void k_atom(const int* __restrict__ x, const float* __restrict__ emb,
                       float* __restrict__ h) {
    int n = blockIdx.x;
    int c = threadIdx.x;
    __shared__ int xr[NFEAT];
    if (c < NFEAT) xr[c] = x[n * NFEAT + c];
    __syncthreads();
    float s = 0.f;
#pragma unroll
    for (int f = 0; f < NFEAT; ++f)
        s += emb[((size_t)f * VOCAB + xr[f]) * HID + c];
    h[(size_t)n * HID + c] = s;
}

// ---------------------------------------------------------------- WMMA GEMM
// C[M,Kout] = A[M,K] * W[K,Kout] (+ bias[col]); f32 in/out, bf16 WMMA compute.
// Block = 128 threads (4 waves). Block stages a bf16 fragment-swizzled panel of
// W (NTG*16 columns x K) into dynamic LDS; each wave computes a 16-row strip
// with NTG accumulators, reusing one A fragment across NTG WMMAs per K-step.
template <int K, int NTG>
__global__ __launch_bounds__(128)
void k_gemm(const float* __restrict__ A, const float* __restrict__ W,
            const float* __restrict__ bias, float* __restrict__ C,
            int Mtiles, int Kout) {
    extern __shared__ __align__(32) char smem[];
    __bf16*       sBh = (__bf16*)smem;
    const v16bf*  sBv = (const v16bf*)smem;

    constexpr int KSTEPS = K / 32;
    const int colbase = blockIdx.x * (NTG * 16);

    // ---- cooperative staging: W panel -> bf16, WMMA B-fragment layout ----
    // linear idx = ((f * 32) + lane) * 16 + e, f = nt*KSTEPS + kstep
    for (int idx = threadIdx.x; idx < NTG * 16 * K; idx += 128) {
        int e    = idx & 15;
        int lane = (idx >> 4) & 31;
        int f    = idx >> 9;
        int kstep = f % KSTEPS;
        int nt    = f / KSTEPS;
        int hi  = lane >> 4;
        int l15 = lane & 15;
        int k   = kstep * 32 + hi * 16 + e;       // contiguous K halves per lane-col
        int col = colbase + nt * 16 + l15;
        sBh[idx] = (__bf16)W[(size_t)k * Kout + col];
    }
    __syncthreads();

    int wave = threadIdx.x >> 5;
    int lane = threadIdx.x & 31;
    int mtile = blockIdx.y * 4 + wave;
    if (mtile >= Mtiles) return;                   // wave-uniform: EXEC stays full

    int hi  = lane >> 4;
    int l15 = lane & 15;
    const float* arow = A + (size_t)(mtile * 16 + l15) * K;
    int koff = hi * 16;

    v8f acc[NTG];
#pragma unroll
    for (int nt = 0; nt < NTG; ++nt) acc[nt] = (v8f){};

#pragma unroll
    for (int kstep = 0; kstep < KSTEPS; ++kstep) {
        int kb = kstep * 32;
        // A fragment: contiguous K halves per lane, K split {0..15}/{16..31}
        const float4* apLo = (const float4*)(arow + kb + koff);
        const float4* apHi = (const float4*)(arow + kb + koff + 8);
        float4 a0 = apLo[0], a1 = apLo[1];
        float4 a2 = apHi[0], a3 = apHi[1];
        v16bf a;
        a[0]=(__bf16)a0.x; a[1]=(__bf16)a0.y; a[2]=(__bf16)a0.z; a[3]=(__bf16)a0.w;
        a[4]=(__bf16)a1.x; a[5]=(__bf16)a1.y; a[6]=(__bf16)a1.z; a[7]=(__bf16)a1.w;
        a[8]=(__bf16)a2.x; a[9]=(__bf16)a2.y; a[10]=(__bf16)a2.z; a[11]=(__bf16)a2.w;
        a[12]=(__bf16)a3.x; a[13]=(__bf16)a3.y; a[14]=(__bf16)a3.z; a[15]=(__bf16)a3.w;
#pragma unroll
        for (int nt = 0; nt < NTG; ++nt) {
            v16bf b = sBv[(nt * KSTEPS + kstep) * 32 + lane];
            acc[nt] = __builtin_amdgcn_wmma_f32_16x16x32_bf16(
                          false, a, false, b, (short)0, acc[nt], false, false);
        }
    }

#pragma unroll
    for (int nt = 0; nt < NTG; ++nt) {
        int col = colbase + nt * 16 + l15;
        float bv = bias ? bias[col] : 0.f;
#pragma unroll
        for (int r = 0; r < 8; ++r) {              // C/D layout: vgpr r -> row r + 8*hi
            int row = mtile * 16 + r + hi * 8;
            C[(size_t)row * Kout + col] = acc[nt][r] + bv;
        }
    }
}

// ---------------------------------------------------------------- aggregation
__global__ void k_deg(const int* __restrict__ dst, float* __restrict__ deg, int E) {
    int e = blockIdx.x * blockDim.x + threadIdx.x;
    if (e < E) atomicAdd(&deg[dst[e]], 1.0f);
}

__global__ void k_scatter(const float* __restrict__ bases,
                          const int* __restrict__ src, const int* __restrict__ dst,
                          float* __restrict__ agg_sum, float* __restrict__ agg_max) {
    int e = blockIdx.x;
    int c = threadIdx.x;                           // 128 threads
    int s = src[e], d = dst[e];
    float v = bases[(size_t)s * BF + c];
    atomicAdd(&agg_sum[(size_t)d * BF + c], v);
    atomicMaxF(&agg_max[(size_t)d * BF + c], v);
}

// ---------------------------------------------------------------- per-node einsum
__global__ void k_combine(const float* __restrict__ agg_sum,
                          const float* __restrict__ agg_max,
                          const float* __restrict__ deg,
                          const float* __restrict__ wc,
                          const float* __restrict__ convBias,
                          float* __restrict__ cout) {
    int n = blockIdx.x;
    int t = threadIdx.x;                           // 256 threads
    __shared__ float sagg[AGGS][BF];
    __shared__ float sw[COMBD];
    if (t < BF) {
        float su = agg_sum[(size_t)n * BF + t];
        float dg = deg[n];
        sagg[0][t] = su;
        sagg[1][t] = su / fmaxf(dg, 1.f);
        sagg[2][t] = agg_max[(size_t)n * BF + t];
    } else if (t < BF + COMBD) {
        sw[t - BF] = wc[(size_t)n * COMBD + (t - BF)];
    }
    __syncthreads();
    int hd = t >> 5, d = t & 31;
    float acc = convBias[t];
#pragma unroll
    for (int k = 0; k < AGGS * BASES; ++k)         // k = a*4 + b
        acc += sw[hd * 12 + k] * sagg[k >> 2][(k & 3) * 32 + d];
    cout[(size_t)n * HID + t] = acc;
}

// ---------------------------------------------------------------- batchnorm
__global__ void k_colstats(const float* __restrict__ X, int Nrows, int C,
                           int rows_per_block, float* __restrict__ s,
                           float* __restrict__ sq) {
    int c = threadIdx.x;                           // blockDim.x == C
    int r0 = blockIdx.x * rows_per_block;
    int r1 = min(r0 + rows_per_block, Nrows);
    float a = 0.f, b = 0.f;
    for (int r = r0; r < r1; ++r) {
        float v = X[(size_t)r * C + c];
        a += v; b += v * v;
    }
    atomicAdd(&s[c], a);
    atomicAdd(&sq[c], b);
}

__global__ void k_bn(const float* __restrict__ Xin, float* __restrict__ Out,
                     const float* __restrict__ s, const float* __restrict__ sq,
                     const float* __restrict__ gamma, const float* __restrict__ beta,
                     int Nrows, int C, int residual) {
    size_t i = (size_t)blockIdx.x * blockDim.x + threadIdx.x;
    size_t total = (size_t)Nrows * C;
    if (i >= total) return;
    int c = (int)(i % C);
    float inv = 1.0f / (float)Nrows;
    float mu  = s[c] * inv;
    float var = sq[c] * inv - mu * mu;
    float v = (Xin[i] - mu) * rsqrtf(var + EPSV) * gamma[c] + beta[c];
    v = fmaxf(v, 0.f);
    Out[i] = residual ? (Out[i] + v) : v;
}

// ---------------------------------------------------------------- pooling
__global__ void k_pool(const float* __restrict__ h, const int* __restrict__ batch,
                       float* __restrict__ gsum, float* __restrict__ cnt) {
    int n = blockIdx.x;
    int c = threadIdx.x;                           // 256
    int g = batch[n];
    atomicAdd(&gsum[(size_t)g * HID + c], h[(size_t)n * HID + c]);
    if (c == 0) atomicAdd(&cnt[g], 1.0f);
}

__global__ void k_pooldiv(const float* __restrict__ gsum, const float* __restrict__ cnt,
                          float* __restrict__ g0) {
    int g = blockIdx.x;
    int c = threadIdx.x;
    g0[(size_t)g * HID + c] = gsum[(size_t)g * HID + c] / fmaxf(cnt[g], 1.f);
}

// ---------------------------------------------------------------- head
__global__ void k_head(const float* __restrict__ g2, const float* __restrict__ w3,
                       const float* __restrict__ b3, float* __restrict__ out) {
    int n = blockIdx.x * blockDim.x + threadIdx.x;
    if (n >= NG) return;
    float a = b3[0];
#pragma unroll
    for (int k = 0; k < 64; ++k)
        a += g2[(size_t)n * 64 + k] * w3[k];
    out[n] = a;
}

// ---------------------------------------------------------------- launch
extern "C" void kernel_launch(void* const* d_in, const int* in_sizes, int n_in,
                              void* d_out, int out_size, void* d_ws, size_t ws_size,
                              hipStream_t stream) {
    const int*   x        = (const int*)  d_in[0];
    const int*   esrc     = (const int*)  d_in[1];
    const int*   edst     = (const int*)  d_in[2];
    const int*   batch    = (const int*)  d_in[3];
    const float* emb      = (const float*)d_in[4];
    const float* basesW   = (const float*)d_in[5];   // [L,256,128]
    const float* combW    = (const float*)d_in[6];   // [L,256,96]
    const float* combB    = (const float*)d_in[7];   // [L,96]
    const float* convBias = (const float*)d_in[8];   // [L,256]
    const float* bnG      = (const float*)d_in[9];   // [L,256]
    const float* bnB      = (const float*)d_in[10];  // [L,256]
    const float* w1       = (const float*)d_in[11];  // [256,128]
    const float* bn1g     = (const float*)d_in[12];
    const float* bn1b     = (const float*)d_in[13];
    const float* w2       = (const float*)d_in[14];  // [128,64]
    const float* bn2g     = (const float*)d_in[15];
    const float* bn2b     = (const float*)d_in[16];
    const float* w3       = (const float*)d_in[17];  // [64,1]
    const float* b3       = (const float*)d_in[18];
    float* out = (float*)d_out;

    const int E = in_sizes[1];                       // 1,700,000

    // ---- carve workspace (all offsets 256B aligned) ----
    char* ws = (char*)d_ws;
    size_t off = 0;
    auto carve = [&](size_t bytes) -> float* {
        float* p = (float*)(ws + off);
        off += (bytes + 255) & ~(size_t)255;
        return p;
    };
    float* h       = carve((size_t)NN * HID * 4);
    float* cbuf    = carve((size_t)NN * HID * 4);
    float* bases   = carve((size_t)NN * BF * 4);
    float* wc      = carve((size_t)NN * COMBD * 4);
    float* agg_sum = carve((size_t)NN * BF * 4);
    float* agg_max = carve((size_t)NN * BF * 4);
    float* deg     = carve((size_t)NN * 4);
    float* ssum    = carve(HID * 4);
    float* ssq     = carve(HID * 4);
    float* gsum    = carve((size_t)NG * HID * 4);
    float* cnt     = carve((size_t)NG * 4);
    const int NGP = 5008;                            // 313 * 16
    float* g0      = carve((size_t)NGP * 256 * 4);
    float* g1      = carve((size_t)NGP * 128 * 4);
    float* g2      = carve((size_t)NGP * 64 * 4);
    (void)ws_size; (void)n_in; (void)out_size;

    const float NEG_INF = -INFINITY;
    dim3 fillGrid(2048), fillBlk(256);

    // ---- atom encoder ----
    k_atom<<<NN, HID, 0, stream>>>(x, emb, h);

    // ---- EGConv layers ----
    const int Mtiles = NN / 16;                      // 6250, exact
    const int mgrid  = (Mtiles + 3) / 4;
    const size_t lds_256x64 = (size_t)256 * 64 * 2;  // 32 KB
    const size_t lds_256x48 = (size_t)256 * 48 * 2;  // 24 KB
    const size_t lds_128x64 = (size_t)128 * 64 * 2;  // 16 KB

    for (int l = 0; l < NLAYER; ++l) {
        // dense projections (WMMA, LDS-staged bf16 weight panels)
        k_gemm<256, 4><<<dim3(2, mgrid), 128, lds_256x64, stream>>>(
            h, basesW + (size_t)l * HID * BF, nullptr, bases, Mtiles, BF);
        k_gemm<256, 3><<<dim3(2, mgrid), 128, lds_256x48, stream>>>(
            h, combW + (size_t)l * HID * COMBD, combB + (size_t)l * COMBD,
            wc, Mtiles, COMBD);

        // reset aggregation buffers
        k_fill<<<fillGrid, fillBlk, 0, stream>>>(agg_sum, (size_t)NN * BF, 0.f);
        k_fill<<<fillGrid, fillBlk, 0, stream>>>(agg_max, (size_t)NN * BF, NEG_INF);
        k_fill<<<fillGrid, fillBlk, 0, stream>>>(deg, (size_t)NN, 0.f);

        // scatter
        k_deg<<<(E + 255) / 256, 256, 0, stream>>>(edst, deg, E);
        k_scatter<<<E, BF, 0, stream>>>(bases, esrc, edst, agg_sum, agg_max);

        // per-node combine (einsum) + conv bias
        k_combine<<<NN, HID, 0, stream>>>(agg_sum, agg_max, deg, wc,
                                          convBias + (size_t)l * HID, cbuf);

        // batchnorm (training-mode stats) + relu + residual
        k_fill<<<1, HID, 0, stream>>>(ssum, HID, 0.f);
        k_fill<<<1, HID, 0, stream>>>(ssq, HID, 0.f);
        k_colstats<<<(NN + 255) / 256, HID, 0, stream>>>(cbuf, NN, HID, 256, ssum, ssq);
        k_bn<<<(int)(((size_t)NN * HID + 255) / 256), 256, 0, stream>>>(
            cbuf, h, ssum, ssq, bnG + (size_t)l * HID, bnB + (size_t)l * HID,
            NN, HID, /*residual=*/1);
    }

    // ---- global mean pool ----
    k_fill<<<fillGrid, fillBlk, 0, stream>>>(gsum, (size_t)NG * HID, 0.f);
    k_fill<<<fillGrid, fillBlk, 0, stream>>>(cnt, (size_t)NG, 0.f);
    k_fill<<<fillGrid, fillBlk, 0, stream>>>(g0, (size_t)NGP * 256, 0.f);  // zero pad rows
    k_pool<<<NN, HID, 0, stream>>>(h, batch, gsum, cnt);
    k_pooldiv<<<NG, HID, 0, stream>>>(gsum, cnt, g0);

    // ---- MLP head (padded rows are zero -> propagate zeros through gemms) ----
    const int MtG = NGP / 16;                        // 313
    const int ggrid = (MtG + 3) / 4;
    k_gemm<256, 4><<<dim3(2, ggrid), 128, lds_256x64, stream>>>(
        g0, w1, nullptr, g1, MtG, 128);
    k_fill<<<1, 128, 0, stream>>>(ssum, 128, 0.f);
    k_fill<<<1, 128, 0, stream>>>(ssq, 128, 0.f);
    k_colstats<<<(NG + 255) / 256, 128, 0, stream>>>(g1, NG, 128, 256, ssum, ssq);
    k_bn<<<(int)(((size_t)NG * 128 + 255) / 256), 256, 0, stream>>>(
        g1, g1, ssum, ssq, bn1g, bn1b, NG, 128, /*residual=*/0);

    k_gemm<128, 4><<<dim3(1, ggrid), 128, lds_128x64, stream>>>(
        g1, w2, nullptr, g2, MtG, 64);
    k_fill<<<1, 64, 0, stream>>>(ssum, 64, 0.f);
    k_fill<<<1, 64, 0, stream>>>(ssq, 64, 0.f);
    k_colstats<<<(NG + 255) / 256, 64, 0, stream>>>(g2, NG, 64, 256, ssum, ssq);
    k_bn<<<(int)(((size_t)NG * 64 + 255) / 256), 256, 0, stream>>>(
        g2, g2, ssum, ssq, bn2g, bn2b, NG, 64, /*residual=*/0);

    k_head<<<(NG + 255) / 256, 256, 0, stream>>>(g2, w3, b3, out);
}
